// TempME_9268539424953
// MI455X (gfx1250) — matrix operand order repaired and checked by reading, not compiled, not run on hardware
//
#include <hip/hip_runtime.h>
#include <hip/hip_bf16.h>
#include <cmath>

typedef __attribute__((ext_vector_type(16))) _Float16 v16h;
typedef __attribute__((ext_vector_type(8)))  _Float16 v8h;
typedef __attribute__((ext_vector_type(8)))  float    v8f;

namespace {
constexpr int kB  = 256;
constexpr int kW  = 128;
constexpr int kL  = 3;
constexpr int kR3 = kB * kW * kL;   // 98304 event rows
constexpr int kR1 = kB * kW;        // 32768 window rows
}

// ---------------------------------------------------------------------------
// Repack a row-major f32 weight [K x N] into WMMA-B tile layout (f16):
//   tiles of 32(K) x 16(N); within a tile: lane L holds column N = L&15,
//   halves e=0..15 hold K = kt*32 + 16*(L>=16) + e.  Zero padding outside K,N.
// ---------------------------------------------------------------------------
__global__ void pack_weight_kernel(const float* __restrict__ src,
                                   _Float16* __restrict__ dst,
                                   int K, int N, int ntiles, int total) {
  int idx = blockIdx.x * blockDim.x + threadIdx.x;
  if (idx >= total) return;
  int tile = idx >> 9;
  int r    = idx & 511;
  int lane = r >> 4;
  int e    = r & 15;
  int kt = tile / ntiles, nt = tile - kt * ntiles;
  int k = kt * 32 + ((lane & 16) ? 16 : 0) + e;
  int n = nt * 16 + (lane & 15);
  float v = (k < K && n < N) ? src[(long)k * N + n] : 0.0f;
  dst[idx] = (_Float16)v;
}

// ---------------------------------------------------------------------------
// Build event matrix [R3 x 352] f16:
//   cols 0..171  = edge_table[edge_idx[row]]
//   cols 172..174= edge_identify[row]
//   cols 175..346= cos(delta * basis_freq + phase),  delta = cut[b] - time[row]
//   cols 347..351= 0 (K padding)
// ---------------------------------------------------------------------------
__global__ void build_event_kernel(const float* __restrict__ edge_table,
                                   const int*   __restrict__ edge_idx,
                                   const float* __restrict__ edge_identify,
                                   const float* __restrict__ time_idx,
                                   const float* __restrict__ cut_time_l,
                                   const float* __restrict__ basis_freq,
                                   const float* __restrict__ phase,
                                   _Float16* __restrict__ event) {
  long idx = (long)blockIdx.x * blockDim.x + threadIdx.x;
  if (idx >= (long)kR3 * 352) return;
  int row = (int)(idx / 352);
  int c   = (int)(idx - (long)row * 352);
  float val = 0.0f;
  if (c < 172) {
    val = edge_table[(long)edge_idx[row] * 172 + c];
  } else if (c < 175) {
    val = edge_identify[(long)row * 3 + (c - 172)];
  } else if (c < 347) {
    int k = c - 175;
    int b = row / (kW * kL);
    float delta = cut_time_l[b] - time_idx[row];
    val = cosf(delta * basis_freq[k] + phase[k]);
  }
  event[idx] = (_Float16)val;
}

// ---------------------------------------------------------------------------
// Epilogues
// ---------------------------------------------------------------------------
struct EpiF32 {               // C[m,n] = acc + bias[n]   (f32 out)
  float* C; const float* bias; int ldc;
  __device__ void apply(int mbase, int n, const v8f& c) const {
    float bv = bias[n];
#pragma unroll
    for (int v = 0; v < 8; ++v)
      C[(long)(mbase + v) * ldc + n] = c[v] + bv;
  }
};

struct EpiF16 {               // f16 out, optional relu, zero past nValid
  _Float16* C; const float* bias; int ldc; int nValid; int relu;
  __device__ void apply(int mbase, int n, const v8f& c) const {
    if (n >= nValid) {
#pragma unroll
      for (int v = 0; v < 8; ++v) C[(long)(mbase + v) * ldc + n] = (_Float16)0.0f;
      return;
    }
    float bv = bias[n];
#pragma unroll
    for (int v = 0; v < 8; ++v) {
      float x = c[v] + bv;
      if (relu) x = fmaxf(x, 0.0f);
      C[(long)(mbase + v) * ldc + n] = (_Float16)x;
    }
  }
};

// ev GEMM epilogue: fuse node-table gathers and both conv pre-activations.
//   ev = acc + ev_b[n];  s = node_table[src]; t = node_table[tgt]
//   u1 = s + relu(t + ev)   (conv(src,tgt) input)
//   u2 = t + relu(s + ev)   (conv(tgt,src) input)
struct EpiEv {
  const float* node_table; const int* node_idx; const float* ev_b;
  _Float16* u1; _Float16* u2;
  __device__ void apply(int mbase, int n, const v8f& c) const {
#pragma unroll
    for (int v = 0; v < 8; ++v) {
      int m = mbase + v;                     // event row
      if (n < 172) {
        int bw = m / 3, l = m - bw * 3;
        int s_id = node_idx[bw * 6 + 2 * l];
        int t_id = node_idx[bw * 6 + 2 * l + 1];
        float ev = c[v] + ev_b[n];
        float s  = node_table[(long)s_id * 172 + n];
        float t  = node_table[(long)t_id * 172 + n];
        u1[(long)m * 192 + n] = (_Float16)(s + fmaxf(t + ev, 0.0f));
        u2[(long)m * 192 + n] = (_Float16)(t + fmaxf(s + ev, 0.0f));
      } else {                               // N padding (172..175)
        u1[(long)m * 192 + n] = (_Float16)0.0f;
        u2[(long)m * 192 + n] = (_Float16)0.0f;
      }
    }
  }
};

// ---------------------------------------------------------------------------
// Generic tiled GEMM: one wave computes one 16x16 output tile.
//   A: row-major f16, row index = rowOff + m*rowMul (affine row map)
//   Bp: pre-packed WMMA-B tiles; K consumed 32/step via v_wmma_f32_16x16x32_f16
// ---------------------------------------------------------------------------
template <class Epi>
__global__ void __launch_bounds__(128)
gemm16_kernel(const _Float16* __restrict__ A, const _Float16* __restrict__ Bp,
              int lda, int ksteps, int ntiles, int totalTiles,
              int rowMul, int rowOff, Epi epi) {
  int wid = (blockIdx.x << 2) + (threadIdx.x >> 5);
  if (wid >= totalTiles) return;               // uniform per-wave: EXEC stays all-1s
  int lane = threadIdx.x & 31;
  int mt = wid / ntiles;
  int nt = wid - mt * ntiles;
  int hi = (lane & 16) ? 1 : 0;
  // A lane layout: M = lane&15; halves 0-15 -> K 16*(lane>=16) + e
  const _Float16* arow =
      A + (long)(rowOff + (mt * 16 + (lane & 15)) * rowMul) * lda + hi * 16;
  const _Float16* bpt = Bp + (long)nt * 512 + lane * 16;
  const int bstride = ntiles * 512;

  v8f c = {};
  for (int ks = 0; ks < ksteps; ++ks) {
    v8h alo = *(const v8h*)(arow);
    v8h ahi = *(const v8h*)(arow + 8);
    v16h a;
#pragma unroll
    for (int i = 0; i < 8; ++i) { a[i] = alo[i]; a[i + 8] = ahi[i]; }
    v16h b = *(const v16h*)(bpt);
    c = __builtin_amdgcn_wmma_f32_16x16x32_f16(false, a, false, b,
                                               (short)0, c, false, false);
    arow += 32;
    bpt += bstride;
    __builtin_prefetch(arow, 0, 0);            // global_prefetch_b8 next A chunk
  }
  int n     = nt * 16 + (lane & 15);
  int mbase = mt * 16 + hi * 8;
  epi.apply(mbase, n, c);
}

// ---------------------------------------------------------------------------
// Global sum / sum-of-squares of diff = |cut[b] - time_idx[b,w,j]|, j<2
// ---------------------------------------------------------------------------
__global__ void diff_reduce_kernel(const float* __restrict__ time_idx,
                                   const float* __restrict__ cut,
                                   float* __restrict__ red) {
  __shared__ float ssum[256];
  __shared__ float ssq[256];
  int tid = threadIdx.x;
  float s = 0.0f, q = 0.0f;
  const int total = kR1 * 2;
  for (int i = blockIdx.x * blockDim.x + tid; i < total;
       i += gridDim.x * blockDim.x) {
    int bw = i >> 1, j = i & 1;
    int b = bw / kW;
    float d = fabsf(cut[b] - time_idx[bw * 3 + j]);
    s += d; q += d * d;
  }
  ssum[tid] = s; ssq[tid] = q;
  __syncthreads();
  for (int st = 128; st > 0; st >>= 1) {
    if (tid < st) { ssum[tid] += ssum[tid + st]; ssq[tid] += ssq[tid + st]; }
    __syncthreads();
  }
  if (tid == 0) { atomicAdd(&red[0], ssum[0]); atomicAdd(&red[1], ssq[0]); }
}

// ---------------------------------------------------------------------------
// Attention: one wave per window row. scores -> time-weight -> softmax(2) ->
//   out[d] = s_feat[d] + a0*Wq0[d] + a1*Wq1[d]   (f16 out for next GEMM)
// ---------------------------------------------------------------------------
__global__ void attention_kernel(const float* __restrict__ Wp,
                                 const float* __restrict__ Wq,
                                 const _Float16* __restrict__ upd,
                                 const float* __restrict__ time_idx,
                                 const float* __restrict__ cut,
                                 const float* __restrict__ red,
                                 _Float16* __restrict__ outf) {
  int wid  = (blockIdx.x * blockDim.x + threadIdx.x) >> 5;
  int lane = threadIdx.x & 31;
  if (wid >= kR1) return;
  int r = wid;
  int b = r / kW;
  const float* wp  = Wp + (long)r * 256;
  const float* wq0 = Wq + (long)(3 * r) * 256;
  const float* wq1 = wq0 + 256;
  float p0 = 0.0f, p1 = 0.0f;
  for (int d = lane; d < 256; d += 32) {
    float pv = wp[d];
    p0 += pv * wq0[d];
    p1 += pv * wq1[d];
  }
  for (int o = 16; o > 0; o >>= 1) {
    p0 += __shfl_xor(p0, o, 32);
    p1 += __shfl_xor(p1, o, 32);
  }
  const float nE = (float)(kR1 * 2);
  float sum = red[0], sq = red[1];
  float var = (sq - sum * sum / nE) / (nE - 1.0f);
  float sd  = sqrtf(fmaxf(var, 0.0f));
  float inv = 1.0f / (sd + 1e-6f);
  float tw0 = expf(-fabsf(cut[b] - time_idx[r * 3 + 0]) * inv);
  float tw1 = expf(-fabsf(cut[b] - time_idx[r * 3 + 1]) * inv);
  float s0 = p0 * (0.7f + 0.3f * tw0);
  float s1 = p1 * (0.7f + 0.3f * tw1);
  float mx = fmaxf(s0, s1);
  float e0 = expf(s0 - mx), e1 = expf(s1 - mx);
  float a0 = e0 / (e0 + e1), a1 = e1 / (e0 + e1);
  const _Float16* sf = upd + (long)(3 * r + 2) * 256;   // s_feat = upd[:,:,2,:]
  _Float16* po = outf + (long)r * 256;
  for (int d = lane; d < 256; d += 32) {
    float ov = (float)sf[d] + a0 * wq0[d] + a1 * wq1[d];
    po[d] = (_Float16)ov;
  }
}

// x[:,128:140] = cat_feat (cols 0..127 filled by at_m2 GEMM; 140..159 memset 0)
__global__ void cat_fill_kernel(const float* __restrict__ cat_feat,
                                _Float16* __restrict__ x) {
  int idx = blockIdx.x * blockDim.x + threadIdx.x;
  if (idx >= kR1 * 12) return;
  int r = idx / 12, c = idx - r * 12;
  x[(long)r * 160 + 128 + c] = (_Float16)cat_feat[idx];
}

// out[r] = z2[r,:] . f_W3 + f_b3
__global__ void final_dot_kernel(const _Float16* __restrict__ z2,
                                 const float* __restrict__ fW3,
                                 const float* __restrict__ fb3,
                                 float* __restrict__ out) {
  int r = blockIdx.x * blockDim.x + threadIdx.x;
  if (r >= kR1) return;
  float acc = fb3[0];
  const _Float16* zr = z2 + (long)r * 128;
#pragma unroll 8
  for (int k = 0; k < 128; ++k) acc += (float)zr[k] * fW3[k];
  out[r] = acc;
}

// ---------------------------------------------------------------------------
extern "C" void kernel_launch(void* const* d_in, const int* in_sizes, int n_in,
                              void* d_out, int out_size, void* d_ws, size_t ws_size,
                              hipStream_t stream) {
  (void)in_sizes; (void)n_in; (void)out_size; (void)ws_size;
  const float* node_table    = (const float*)d_in[0];
  const float* edge_table    = (const float*)d_in[1];
  const float* basis_freq    = (const float*)d_in[2];
  const float* phase         = (const float*)d_in[3];
  const float* ev_W          = (const float*)d_in[4];
  const float* ev_b          = (const float*)d_in[5];
  const float* ev_m1W        = (const float*)d_in[6];
  const float* ev_m1b        = (const float*)d_in[7];
  const float* ev_m2W        = (const float*)d_in[8];
  const float* ev_m2b        = (const float*)d_in[9];
  const float* at_W1         = (const float*)d_in[10];
  const float* at_b1         = (const float*)d_in[11];
  const float* at_W2         = (const float*)d_in[12];
  const float* at_b2         = (const float*)d_in[13];
  const float* at_m1W        = (const float*)d_in[14];
  const float* at_m1b        = (const float*)d_in[15];
  const float* at_m2W        = (const float*)d_in[16];
  const float* at_m2b        = (const float*)d_in[17];
  const float* f_W1          = (const float*)d_in[18];
  const float* f_b1          = (const float*)d_in[19];
  const float* f_W2          = (const float*)d_in[20];
  const float* f_b2          = (const float*)d_in[21];
  const float* f_W3          = (const float*)d_in[22];
  const float* f_b3          = (const float*)d_in[23];
  const float* time_idx      = (const float*)d_in[24];
  const float* cut_time_l    = (const float*)d_in[25];
  const float* cat_feat      = (const float*)d_in[26];
  const float* edge_identify = (const float*)d_in[27];
  const int*   node_idx      = (const int*)d_in[28];
  const int*   edge_idx      = (const int*)d_in[29];

  char* ws = (char*)d_ws;

  // ---- workspace arena (overlapped lifetimes) ----
  const size_t OFF_EVENT = 0;                              // f16 [R3 x 352] (dead after ev GEMM)
  const size_t OFF_WQ    = 0;                              // f32 [R3 x 256] (reuses event region)
  const size_t OFF_U1    = (size_t)kR3 * 256 * 4;          // f16 [R3 x 192]
  const size_t SZ_U      = (size_t)kR3 * 192 * 2;
  const size_t OFF_U2    = OFF_U1 + SZ_U;                  // f16 [R3 x 192]
  const size_t OFF_H1    = OFF_U2 + SZ_U;                  // f16 [R3 x 128]
  const size_t OFF_UPD   = OFF_H1 + (size_t)kR3 * 128 * 2; // f16 [R3 x 256]
  const size_t OFF_WP    = OFF_UPD + (size_t)kR3 * 256 * 2;// f32 [R1 x 256]
  const size_t WS_HI     = OFF_WP + (size_t)kR1 * 256 * 4;
  // tail buffers overlap dead u1/u2 region:
  const size_t OFF_OUTF  = OFF_U1;                         // f16 [R1 x 256]
  const size_t OFF_HA    = OFF_OUTF + (size_t)kR1 * 256 * 2; // f16 [R1 x 128]
  const size_t OFF_X     = OFF_HA + (size_t)kR1 * 128 * 2;   // f16 [R1 x 160]
  const size_t OFF_Z1    = OFF_X + (size_t)kR1 * 160 * 2;    // f16 [R1 x 160]
  const size_t OFF_Z2    = OFF_Z1 + (size_t)kR1 * 160 * 2;   // f16 [R1 x 128]
  // packed weights + reduction scalars above WS_HI:
  size_t po = WS_HI;
  auto palloc = [&](int kt, int nt) {
    size_t o = po; po += (size_t)kt * nt * 512 * 2; po = (po + 255) & ~(size_t)255;
    return o;
  };
  const size_t P_EV   = palloc(11, 11);
  const size_t P_M1   = palloc(6, 8);
  const size_t P_M2   = palloc(4, 8);
  const size_t P_AT1  = palloc(8, 16);
  const size_t P_AT2  = palloc(8, 16);
  const size_t P_ATM1 = palloc(8, 8);
  const size_t P_ATM2 = palloc(4, 8);
  const size_t P_F1   = palloc(5, 9);
  const size_t P_F2   = palloc(5, 8);
  const size_t OFF_RED = po;

  auto pack = [&](const float* src, size_t dst, int K, int N, int kt, int nt) {
    int total = kt * nt * 512;
    pack_weight_kernel<<<(total + 255) / 256, 256, 0, stream>>>(
        src, (_Float16*)(ws + dst), K, N, nt, total);
  };
  pack(ev_W,   P_EV,   347, 172, 11, 11);
  pack(ev_m1W, P_M1,   172, 128,  6,  8);
  pack(ev_m2W, P_M2,   128, 128,  4,  8);
  pack(at_W1,  P_AT1,  256, 256,  8, 16);
  pack(at_W2,  P_AT2,  256, 256,  8, 16);
  pack(at_m1W, P_ATM1, 256, 128,  8,  8);
  pack(at_m2W, P_ATM2, 128, 128,  4,  8);
  pack(f_W1,   P_F1,   140, 140,  5,  9);
  pack(f_W2,   P_F2,   140, 128,  5,  8);

  hipMemsetAsync(ws + OFF_U1, 0, 2 * SZ_U, stream);  // zero u1,u2 (K-pad cols)
  hipMemsetAsync(ws + OFF_RED, 0, 8, stream);

  auto blocks4 = [](int tiles) { return (tiles + 3) / 4; };
  const int MT3 = kR3 / 16;   // 6144
  const int MT1 = kR1 / 16;   // 2048

  // 1) build event matrix
  {
    long total = (long)kR3 * 352;
    build_event_kernel<<<(int)((total + 255) / 256), 256, 0, stream>>>(
        edge_table, edge_idx, edge_identify, time_idx, cut_time_l,
        basis_freq, phase, (_Float16*)(ws + OFF_EVENT));
  }
  // 2) ev GEMM (fused gathers + both conv inputs)
  {
    EpiEv epi{node_table, node_idx, ev_b,
              (_Float16*)(ws + OFF_U1), (_Float16*)(ws + OFF_U2)};
    int tiles = MT3 * 11;
    gemm16_kernel<<<blocks4(tiles), 128, 0, stream>>>(
        (const _Float16*)(ws + OFF_EVENT), (const _Float16*)(ws + P_EV),
        352, 11, 11, tiles, 1, 0, epi);
  }
  // 3) conv(src,tgt): m1 (relu) -> h1 ; m2 -> upd cols 0..127
  {
    EpiF16 epi{(_Float16*)(ws + OFF_H1), ev_m1b, 128, 128, 1};
    int tiles = MT3 * 8;
    gemm16_kernel<<<blocks4(tiles), 128, 0, stream>>>(
        (const _Float16*)(ws + OFF_U1), (const _Float16*)(ws + P_M1),
        192, 6, 8, tiles, 1, 0, epi);
  }
  {
    EpiF16 epi{(_Float16*)(ws + OFF_UPD), ev_m2b, 256, 128, 0};
    int tiles = MT3 * 8;
    gemm16_kernel<<<blocks4(tiles), 128, 0, stream>>>(
        (const _Float16*)(ws + OFF_H1), (const _Float16*)(ws + P_M2),
        128, 4, 8, tiles, 1, 0, epi);
  }
  // 4) conv(tgt,src): m1 -> h1 ; m2 -> upd cols 128..255
  {
    EpiF16 epi{(_Float16*)(ws + OFF_H1), ev_m1b, 128, 128, 1};
    int tiles = MT3 * 8;
    gemm16_kernel<<<blocks4(tiles), 128, 0, stream>>>(
        (const _Float16*)(ws + OFF_U2), (const _Float16*)(ws + P_M1),
        192, 6, 8, tiles, 1, 0, epi);
  }
  {
    EpiF16 epi{(_Float16*)(ws + OFF_UPD) + 128, ev_m2b, 256, 128, 0};
    int tiles = MT3 * 8;
    gemm16_kernel<<<blocks4(tiles), 128, 0, stream>>>(
        (const _Float16*)(ws + OFF_H1), (const _Float16*)(ws + P_M2),
        128, 4, 8, tiles, 1, 0, epi);
  }
  // 5) Wp = s_feat @ at_W1 + b1  (affine row map: row = 2 + 3m)
  {
    EpiF32 epi{(float*)(ws + OFF_WP), at_b1, 256};
    int tiles = MT1 * 16;
    gemm16_kernel<<<blocks4(tiles), 128, 0, stream>>>(
        (const _Float16*)(ws + OFF_UPD), (const _Float16*)(ws + P_AT1),
        256, 8, 16, tiles, 3, 2, epi);
  }
  // 6) Wq(all l) = upd @ at_W2 + b2  (reuses event region)
  {
    EpiF32 epi{(float*)(ws + OFF_WQ), at_b2, 256};
    int tiles = MT3 * 16;
    gemm16_kernel<<<blocks4(tiles), 128, 0, stream>>>(
        (const _Float16*)(ws + OFF_UPD), (const _Float16*)(ws + P_AT2),
        256, 8, 16, tiles, 1, 0, epi);
  }
  // u2 now dead: zero tail buffers that overlap it
  hipMemsetAsync(ws + OFF_X, 0, (size_t)kR1 * 160 * 2, stream);
  hipMemsetAsync(ws + OFF_Z1, 0, (size_t)kR1 * 160 * 2, stream);
  cat_fill_kernel<<<(kR1 * 12 + 255) / 256, 256, 0, stream>>>(
      cat_feat, (_Float16*)(ws + OFF_X));

  // 7) global std(diff) reduction, then attention
  diff_reduce_kernel<<<256, 256, 0, stream>>>(time_idx, cut_time_l,
                                              (float*)(ws + OFF_RED));
  attention_kernel<<<kR1 / 8, 256, 0, stream>>>(
      (const float*)(ws + OFF_WP), (const float*)(ws + OFF_WQ),
      (const _Float16*)(ws + OFF_UPD), time_idx, cut_time_l,
      (const float*)(ws + OFF_RED), (_Float16*)(ws + OFF_OUTF));

  // 8) at_m1 (relu) -> hA ; at_m2 -> x cols 0..127
  {
    EpiF16 epi{(_Float16*)(ws + OFF_HA), at_m1b, 128, 128, 1};
    int tiles = MT1 * 8;
    gemm16_kernel<<<blocks4(tiles), 128, 0, stream>>>(
        (const _Float16*)(ws + OFF_OUTF), (const _Float16*)(ws + P_ATM1),
        256, 8, 8, tiles, 1, 0, epi);
  }
  {
    EpiF16 epi{(_Float16*)(ws + OFF_X), at_m2b, 160, 128, 0};
    int tiles = MT1 * 8;
    gemm16_kernel<<<blocks4(tiles), 128, 0, stream>>>(
        (const _Float16*)(ws + OFF_HA), (const _Float16*)(ws + P_ATM2),
        128, 4, 8, tiles, 1, 0, epi);
  }
  // 9) final MLP: f1 (relu) -> z1 ; f2 (relu) -> z2 ; dot with f_W3
  {
    EpiF16 epi{(_Float16*)(ws + OFF_Z1), f_b1, 160, 140, 1};
    int tiles = MT1 * 9;
    gemm16_kernel<<<blocks4(tiles), 128, 0, stream>>>(
        (const _Float16*)(ws + OFF_X), (const _Float16*)(ws + P_F1),
        160, 5, 9, tiles, 1, 0, epi);
  }
  {
    EpiF16 epi{(_Float16*)(ws + OFF_Z2), f_b2, 128, 128, 1};
    int tiles = MT1 * 8;
    gemm16_kernel<<<blocks4(tiles), 128, 0, stream>>>(
        (const _Float16*)(ws + OFF_Z1), (const _Float16*)(ws + P_F2),
        160, 5, 8, tiles, 1, 0, epi);
  }
  final_dot_kernel<<<kR1 / 256, 256, 0, stream>>>(
      (const _Float16*)(ws + OFF_Z2), f_W3, f_b3, (float*)d_out);
}
